// BasicBlockAsGNN_9663676416897
// MI455X (gfx1250) — compile-verified, matrix operand order checked by reading, not run on hardware
//
#include <hip/hip_runtime.h>
#include <hip/hip_bf16.h>

#ifndef __has_builtin
#define __has_builtin(x) 0
#endif

// ---------- problem constants ----------
#define NB    128          // batch
#define CIN1  64
#define HIN1  56
#define WIN1  56
#define COUT  128
#define HO    28
#define WO    28
#define SPAT  (HO * WO)            // 784
#define NTOT  (NB * SPAT)          // 100352 (GEMM N dimension)
#define K1    576                  // 64*3*3
#define K2    1152                 // 128*3*3
#define KD    64                   // 64*1*1

typedef __attribute__((ext_vector_type(8)))  float  v8f;
typedef __attribute__((ext_vector_type(8)))  __bf16 v8bf;
typedef __attribute__((ext_vector_type(16))) __bf16 v16bf;

// 128-bit payload type for the async-LDS builtin (matches its declared params:
// 'int __vector_size(16) __device__ *' / AS(3) counterpart)
typedef int v4i_gnu __attribute__((vector_size(16)));
typedef __attribute__((address_space(1))) v4i_gnu gv4;
typedef __attribute__((address_space(3))) v4i_gnu lv4;

__device__ __forceinline__ unsigned short f32_to_bf16_bits(float f) {
    unsigned int u = __float_as_uint(f);
    unsigned int lsb = (u >> 16) & 1u;
    u += 0x7fffu + lsb;                 // round-to-nearest-even
    return (unsigned short)(u >> 16);
}

// ---------- gfx1250 async global->LDS (ASYNCcnt-tracked) ----------
__device__ __forceinline__ void async_ld_b128(const unsigned short* g, unsigned short* l) {
#if __has_builtin(__builtin_amdgcn_global_load_async_to_lds_b128)
    __builtin_amdgcn_global_load_async_to_lds_b128(
        (gv4*)(size_t)g, (lv4*)(size_t)l, 0, 0);
#else
    unsigned lofs = (unsigned)(size_t)l;                   // LDS byte address VGPR
    unsigned long long ga = (unsigned long long)(size_t)g; // 64-bit global addr VGPR pair
    asm volatile("global_load_async_to_lds_b128 %0, %1, off"
                 :: "v"(lofs), "v"(ga) : "memory");
#endif
}

__device__ __forceinline__ void wait_async_le2() {
#if __has_builtin(__builtin_amdgcn_s_wait_asynccnt)
    __builtin_amdgcn_s_wait_asynccnt(2);
#else
    asm volatile("s_wait_asynccnt 2" ::: "memory");
#endif
}
__device__ __forceinline__ void wait_async_0() {
#if __has_builtin(__builtin_amdgcn_s_wait_asynccnt)
    __builtin_amdgcn_s_wait_asynccnt(0);
#else
    asm volatile("s_wait_asynccnt 0" ::: "memory");
#endif
}

// ---------- elementwise f32 -> bf16 bits ----------
__global__ __launch_bounds__(256)
void to_bf16_kernel(const float* __restrict__ in, unsigned short* __restrict__ out, long n) {
    long i = (long)blockIdx.x * 256 + threadIdx.x;
    if (i < n) out[i] = f32_to_bf16_bits(in[i]);
}

// ---------- zero the BN stats area ----------
__global__ __launch_bounds__(256)
void zero_stats_kernel(float* __restrict__ stats) {
    for (int i = threadIdx.x; i < 768; i += 256) stats[i] = 0.0f;
}

// ---------- implicit-GEMM conv with WMMA bf16 ----------
// C[m][n] = sum_k W[m][k] * im2col(inp)[k][n]
// block: 256 threads (8 wave32). Tile: M=128 (all Cout), N=64, K-step=32.
// Weights tile: double-buffered LDS fed by GLOBAL_LOAD_ASYNC_TO_LDS_B128.
// im2col tile: register-staged gather -> LDS (transposed [n][k]).
template<int KH, int KW, int STRIDE, int PAD, int KDIM, int CIN, int HIN, int WIN>
__global__ __launch_bounds__(256)
void conv_wmma_kernel(const unsigned short* __restrict__ inp,   // bf16 bits, NCHW
                      const unsigned short* __restrict__ wgt,   // bf16 bits, [COUT][KDIM]
                      float* __restrict__ outp)                 // f32, [B][COUT][HO][WO]
{
    __shared__ unsigned short Alds[2][128][32];  // weights tiles (double buffer)
    __shared__ unsigned short Blds[64][32];      // im2col tile, transposed (N x K)

    const int t    = threadIdx.x;
    const int lane = t & 31;
    const int wave = t >> 5;
    const int n0   = blockIdx.x * 64;

    v8f acc[4];
    #pragma unroll
    for (int j = 0; j < 4; ++j)
        #pragma unroll
        for (int v = 0; v < 8; ++v) acc[j][v] = 0.0f;

    // cooperative A-load mapping: 128 rows x 32 k, 16 bf16 per thread (2 x b128 async)
    const int am  = t >> 1;
    const int ak0 = (t & 1) * 16;

    // cooperative B-load mapping: 64 n x 32 k, 8 bf16 per thread
    const int bn     = t & 63;
    const int bk0    = (t >> 6) * 8;
    const int ncol   = n0 + bn;
    const int bb     = ncol / SPAT;
    const int sp     = ncol - bb * SPAT;
    const int oh     = sp / WO;
    const int ow     = sp - oh * WO;
    const int ihbase = oh * STRIDE - PAD;
    const int iwbase = ow * STRIDE - PAD;

    // WMMA fragment addressing (16-bit A layout, ISA 7.12.2):
    // lanes 0-15: K {0..7,16..23}; lanes 16-31: K {8..15,24..31}
    const int kfrag = (lane & 16) ? 8 : 0;
    const int arow  = wave * 16 + (lane & 15);

    constexpr int NT = KDIM / 32;     // 18 / 36 / 2

    // ---- prologue: async-stage A(0), register-stage B(0) ----
    async_ld_b128(wgt + (size_t)am * KDIM + ak0,     &Alds[0][am][ak0]);
    async_ld_b128(wgt + (size_t)am * KDIM + ak0 + 8, &Alds[0][am][ak0 + 8]);

    unsigned short breg[8];
    #pragma unroll
    for (int i = 0; i < 8; ++i) {
        constexpr int KHKW = KH * KW;
        int k  = bk0 + i;
        int ci = k / KHKW;
        int r  = k - ci * KHKW;
        int kh = r / KW;
        int kw = r - kh * KW;
        int ih = ihbase + kh;
        int iw = iwbase + kw;
        unsigned short v = 0;
        if (ih >= 0 && ih < HIN && iw >= 0 && iw < WIN)
            v = inp[(((size_t)bb * CIN + ci) * HIN + ih) * WIN + iw];
        breg[i] = v;
    }

    for (int it = 0; it < NT; ++it) {
        const int buf = it & 1;
        __syncthreads();   // all waves finished reading LDS from previous step

        // async-stage next A tile into the other buffer
        if (it + 1 < NT) {
            const size_t kb1 = (size_t)(it + 1) * 32;
            async_ld_b128(wgt + (size_t)am * KDIM + kb1 + ak0,     &Alds[buf ^ 1][am][ak0]);
            async_ld_b128(wgt + (size_t)am * KDIM + kb1 + ak0 + 8, &Alds[buf ^ 1][am][ak0 + 8]);
        }

        // commit current B regs to LDS, then gather next B tile into regs
        *reinterpret_cast<uint4*>(&Blds[bn][bk0]) = *reinterpret_cast<const uint4*>(&breg[0]);

        unsigned short bnext[8];
        if (it + 1 < NT) {
            const int kb1 = (it + 1) * 32;
            #pragma unroll
            for (int i = 0; i < 8; ++i) {
                constexpr int KHKW = KH * KW;
                int k  = kb1 + bk0 + i;
                int ci = k / KHKW;
                int r  = k - ci * KHKW;
                int kh = r / KW;
                int kw = r - kh * KW;
                int ih = ihbase + kh;
                int iw = iwbase + kw;
                unsigned short v = 0;
                if (ih >= 0 && ih < HIN && iw >= 0 && iw < WIN)
                    v = inp[(((size_t)bb * CIN + ci) * HIN + ih) * WIN + iw];
                bnext[i] = v;
            }
        }

        // retire this tile's async ops (<=2 leaves next tile's 2 in flight)
        if (it + 1 < NT) wait_async_le2(); else wait_async_0();
        __syncthreads();   // A(it) + B(it) visible to all waves

        // ---- 4 WMMAs per wave per K-step ----
        v8bf alo = *reinterpret_cast<const v8bf*>(&Alds[buf][arow][kfrag]);
        v8bf ahi = *reinterpret_cast<const v8bf*>(&Alds[buf][arow][kfrag + 16]);
        v16bf afrag = __builtin_shufflevector(alo, ahi,
            0,1,2,3,4,5,6,7,8,9,10,11,12,13,14,15);
        #pragma unroll
        for (int j = 0; j < 4; ++j) {
            int brow = j * 16 + (lane & 15);
            v8bf blo = *reinterpret_cast<const v8bf*>(&Blds[brow][kfrag]);
            v8bf bhi = *reinterpret_cast<const v8bf*>(&Blds[brow][kfrag + 16]);
            v16bf bfrag = __builtin_shufflevector(blo, bhi,
                0,1,2,3,4,5,6,7,8,9,10,11,12,13,14,15);
            acc[j] = __builtin_amdgcn_wmma_f32_16x16x32_bf16(
                false, afrag, false, bfrag, (short)0, acc[j], false, false);
        }

        #pragma unroll
        for (int i = 0; i < 8; ++i) breg[i] = bnext[i];
    }

    // ---- epilogue: C/D layout (VGPR v: lanes0-15 M=v, lanes16-31 M=v+8) ----
    const int m0  = wave * 16;
    const int rlo = (lane & 16) ? 8 : 0;
    #pragma unroll
    for (int j = 0; j < 4; ++j) {
        int col = n0 + j * 16 + (lane & 15);
        int b   = col / SPAT;
        int s   = col - b * SPAT;
        #pragma unroll
        for (int v = 0; v < 8; ++v) {
            int row = m0 + rlo + v;
            outp[((size_t)b * COUT + row) * SPAT + s] = acc[j][v];
        }
    }
}

// ---------- per-channel sum / sumsq over (N,H,W) ----------
__global__ __launch_bounds__(256)
void bn_stats_kernel(const float* __restrict__ data,
                     float* __restrict__ sums, float* __restrict__ sumsqs)
{
    __shared__ float ssum[256];
    __shared__ float ssq[256];
    const int c      = blockIdx.x;       // channel
    const int bchunk = blockIdx.y;       // 16 chunks of 8 batches
    float s = 0.0f, q = 0.0f;
    for (int i = threadIdx.x; i < 8 * SPAT; i += 256) {
        int b  = bchunk * 8 + i / SPAT;
        int sp = i % SPAT;
        float v = data[((size_t)b * COUT + c) * SPAT + sp];
        s += v; q += v * v;
    }
    ssum[threadIdx.x] = s; ssq[threadIdx.x] = q;
    __syncthreads();
    for (int off = 128; off > 0; off >>= 1) {
        if (threadIdx.x < off) {
            ssum[threadIdx.x] += ssum[threadIdx.x + off];
            ssq[threadIdx.x]  += ssq[threadIdx.x + off];
        }
        __syncthreads();
    }
    if (threadIdx.x == 0) {
        atomicAdd(&sums[c],   ssum[0]);
        atomicAdd(&sumsqs[c], ssq[0]);
    }
}

// ---------- BN(train) + ReLU -> bf16 activations ----------
__global__ __launch_bounds__(256)
void bn_relu_bf16_kernel(const float* __restrict__ x,
                         const float* __restrict__ sums, const float* __restrict__ sqs,
                         const float* __restrict__ gamma, const float* __restrict__ beta,
                         unsigned short* __restrict__ out)
{
    size_t idx = (size_t)blockIdx.x * 256 + threadIdx.x;      // NB*COUT*SPAT elems
    int c = (int)((idx / SPAT) % COUT);
    const float invN = 1.0f / (float)NTOT;
    float mean = sums[c] * invN;
    float var  = sqs[c] * invN - mean * mean;
    float inv  = rsqrtf(var + 1e-5f);
    float v = (x[idx] - mean) * inv * gamma[c] + beta[c];
    v = fmaxf(v, 0.0f);
    out[idx] = f32_to_bf16_bits(v);
}

// ---------- final: relu(BN2(conv2) + BNd(identity)) ----------
__global__ __launch_bounds__(256)
void bn_add_relu_kernel(const float* __restrict__ c2, const float* __restrict__ idn,
                        const float* __restrict__ sum2, const float* __restrict__ sq2,
                        const float* __restrict__ g2,  const float* __restrict__ b2,
                        const float* __restrict__ sumd, const float* __restrict__ sqd,
                        const float* __restrict__ gd,  const float* __restrict__ bd,
                        float* __restrict__ out)
{
    size_t idx = (size_t)blockIdx.x * 256 + threadIdx.x;
    int c = (int)((idx / SPAT) % COUT);
    const float invN = 1.0f / (float)NTOT;

    float m2 = sum2[c] * invN;
    float v2 = sq2[c] * invN - m2 * m2;
    float a  = (c2[idx] - m2) * rsqrtf(v2 + 1e-5f) * g2[c] + b2[c];

    float md = sumd[c] * invN;
    float vd = sqd[c] * invN - md * md;
    float b  = (idn[idx] - md) * rsqrtf(vd + 1e-5f) * gd[c] + bd[c];

    out[idx] = fmaxf(a + b, 0.0f);
}

// ---------- host-side orchestration ----------
extern "C" void kernel_launch(void* const* d_in, const int* in_sizes, int n_in,
                              void* d_out, int out_size, void* d_ws, size_t ws_size,
                              hipStream_t stream) {
    const float* x  = (const float*)d_in[0];
    const float* w1 = (const float*)d_in[1];
    const float* g1 = (const float*)d_in[2];
    const float* b1 = (const float*)d_in[3];
    const float* w2 = (const float*)d_in[4];
    const float* g2 = (const float*)d_in[5];
    const float* b2 = (const float*)d_in[6];
    const float* wd = (const float*)d_in[7];
    const float* gd = (const float*)d_in[8];
    const float* bd = (const float*)d_in[9];
    float* out = (float*)d_out;

    // workspace layout (all sizes multiples of 256B)
    constexpr size_t N_X    = (size_t)NB * CIN1 * HIN1 * WIN1;   // 25,690,112
    constexpr size_t N_OUT  = (size_t)NB * COUT * SPAT;          // 12,845,056
    char* ws = (char*)d_ws;
    size_t off = 0;
    unsigned short* xbf   = (unsigned short*)(ws + off); off += N_X * 2;
    unsigned short* w1bf  = (unsigned short*)(ws + off); off += (size_t)COUT * K1 * 2;
    unsigned short* w2bf  = (unsigned short*)(ws + off); off += (size_t)COUT * K2 * 2;
    unsigned short* wdbf  = (unsigned short*)(ws + off); off += (size_t)COUT * KD * 2;
    unsigned short* a1bf  = (unsigned short*)(ws + off); off += N_OUT * 2;
    float* convf          = (float*)(ws + off);          off += N_OUT * 4;  // conv1 then conv2
    float* idf            = (float*)(ws + off);          off += N_OUT * 4;
    float* stats          = (float*)(ws + off);          off += 768 * 4;
    float* sum1 = stats;       float* sq1 = stats + 128;
    float* sum2 = stats + 256; float* sq2 = stats + 384;
    float* sumd = stats + 512; float* sqd = stats + 640;

    // 1) precision conversion
    to_bf16_kernel<<<(unsigned)((N_X + 255) / 256), 256, 0, stream>>>(x,  xbf,  (long)N_X);
    to_bf16_kernel<<<(COUT * K1 + 255) / 256, 256, 0, stream>>>(w1, w1bf, COUT * K1);
    to_bf16_kernel<<<(COUT * K2 + 255) / 256, 256, 0, stream>>>(w2, w2bf, COUT * K2);
    to_bf16_kernel<<<(COUT * KD + 255) / 256, 256, 0, stream>>>(wd, wdbf, COUT * KD);
    zero_stats_kernel<<<1, 256, 0, stream>>>(stats);

    const unsigned gemm_grid = NTOT / 64;               // 1568
    const unsigned ew_grid   = (unsigned)(N_OUT / 256); // 50176

    // 2) conv1: 3x3 s2 SAME (pad_begin=0, pad_end via bounds), K=576
    conv_wmma_kernel<3, 3, 2, 0, K1, CIN1, HIN1, WIN1>
        <<<gemm_grid, 256, 0, stream>>>(xbf, w1bf, convf);
    bn_stats_kernel<<<dim3(COUT, 16), 256, 0, stream>>>(convf, sum1, sq1);
    bn_relu_bf16_kernel<<<ew_grid, 256, 0, stream>>>(convf, sum1, sq1, g1, b1, a1bf);

    // 3) conv2: 3x3 s1, pad=1, K=1152 (reuses convf)
    conv_wmma_kernel<3, 3, 1, 1, K2, COUT, HO, WO>
        <<<gemm_grid, 256, 0, stream>>>(a1bf, w2bf, convf);

    // 4) downsample: 1x1 s2 VALID, K=64
    conv_wmma_kernel<1, 1, 2, 0, KD, CIN1, HIN1, WIN1>
        <<<gemm_grid, 256, 0, stream>>>(xbf, wdbf, idf);

    // 5) stats for conv2 / identity
    bn_stats_kernel<<<dim3(COUT, 16), 256, 0, stream>>>(convf, sum2, sq2);
    bn_stats_kernel<<<dim3(COUT, 16), 256, 0, stream>>>(idf,   sumd, sqd);

    // 6) fused BN + BN + add + ReLU -> output
    bn_add_relu_kernel<<<ew_grid, 256, 0, stream>>>(
        convf, idf, sum2, sq2, g2, b2, sumd, sqd, gd, bd, out);

    (void)in_sizes; (void)n_in; (void)out_size; (void)ws_size;
}
